// ObjectAttentionBlock_83829171683582
// MI455X (gfx1250) — matrix operand order repaired and checked by reading, not compiled, not run on hardware
//
#include <hip/hip_runtime.h>
#include <hip/hip_bf16.h>

typedef __attribute__((ext_vector_type(16))) __bf16 v16bf;
typedef __attribute__((ext_vector_type(8)))  __bf16 v8bf;
typedef __attribute__((ext_vector_type(8)))  float  v8f;

// ---------------------------------------------------------------------------
// WMMA helpers (CDNA5: D = A(16x32 bf16) * B(32x16 bf16) + C(16x16 f32))
// ---------------------------------------------------------------------------
static __device__ inline v8f wmma_bf16(v16bf a, v16bf b, v8f c) {
  // 8 args: (neg_a, A, neg_b, B, c_mod, C, reuse_a, reuse_b)
  return __builtin_amdgcn_wmma_f32_16x16x32_bf16(false, a, false, b, (short)0, c, false, false);
}

// A-fragment 16x32 bf16 ISA layout: lane<16 holds K chunks [0..7],[16..23] of
// row M=lane; lane>=16 holds [8..15],[24..31] of row M=lane-16.
// Caller passes c0 = kstep*32 + (lane>>4)*8 ; chunks at c0 and c0+16.
static __device__ inline v16bf a_frag_from_f32(const float* __restrict__ row, int c0) {
  v8f lo = *(const v8f*)(row + c0);
  v8f hi = *(const v8f*)(row + c0 + 16);
  v16bf r;
#pragma unroll
  for (int i = 0; i < 8; ++i) { r[i] = (__bf16)lo[i]; r[i + 8] = (__bf16)hi[i]; }
  return r;
}

static __device__ inline v16bf a_frag_from_lds(const __bf16* base, int m, int stride, int c0) {
  v8bf lo = *(const v8bf*)(base + m * stride + c0);
  v8bf hi = *(const v8bf*)(base + m * stride + c0 + 16);
  return __builtin_shufflevector(lo, hi, 0, 1, 2, 3, 4, 5, 6, 7, 8, 9, 10, 11, 12, 13, 14, 15);
}

// B-fragment 32x16 bf16: lane<16 -> col n=lane, K [0..7],[16..23]; lane>=16 -> K [8..15],[24..31].
// Weights are stored [n][k] bf16; caller passes p at kstep*32 + (lane>>4)*16.
static __device__ inline v16bf b_frag(const __bf16* __restrict__ p) {
  const __bf16* q = p - ((threadIdx.x & 31) >> 4) * 8;
  v8bf lo = *(const v8bf*)q;
  v8bf hi = *(const v8bf*)(q + 16);
  return __builtin_shufflevector(lo, hi, 0, 1, 2, 3, 4, 5, 6, 7, 8, 9, 10, 11, 12, 13, 14, 15);
}

// ---------------------------------------------------------------------------
// Pre-pass kernels (tiny cost; produce bf16 operands in workspace)
// ---------------------------------------------------------------------------
__global__ void k_transpose_bf16(const float* __restrict__ src, __bf16* __restrict__ dst,
                                 int R, int Ccols) {
  int i = blockIdx.x * blockDim.x + threadIdx.x;
  if (i >= R * Ccols) return;
  int r = i / Ccols, c = i % Ccols;
  dst[c * R + r] = (__bf16)src[i];
}

// out[m][n] = relu((sum_k in[m][k]*w[k][n]) * s[n] + b[n]), f32 out
__global__ void k_cbr_naive(const float* __restrict__ in, const float* __restrict__ w,
                            const float* __restrict__ s, const float* __restrict__ b,
                            float* __restrict__ out, int M, int K, int N) {
  int i = blockIdx.x * blockDim.x + threadIdx.x;
  if (i >= M * N) return;
  int m = i / N, n = i % N;
  float a = 0.f;
  for (int k = 0; k < K; ++k) a += in[m * K + k] * w[k * N + n];
  a = a * s[n] + b[n];
  out[i] = a > 0.f ? a : 0.f;
}

// kbf[b][n][f] (n padded to 160, pad rows zero) = cbr(t1 @ kw1)
__global__ void k_class_key(const float* __restrict__ t1, const float* __restrict__ w,
                            const float* __restrict__ s, const float* __restrict__ bb,
                            __bf16* __restrict__ kbf) {
  int i = blockIdx.x * blockDim.x + threadIdx.x;
  if (i >= 8 * 160 * 256) return;
  int f = i % 256, row = i / 256, n = row % 160, b = row / 160;
  if (n >= 150) { kbf[i] = (__bf16)0.f; return; }
  const float* tr = t1 + (b * 150 + n) * 256;
  float a = 0.f;
  for (int k = 0; k < 256; ++k) a += tr[k] * w[k * 256 + f];
  a = a * s[f] + bb[f];
  kbf[i] = (__bf16)(a > 0.f ? a : 0.f);
}

// vT[b][f][n] (n padded to 160, pads zero) = cbr(cc @ vw) transposed
__global__ void k_class_value(const float* __restrict__ cc, const float* __restrict__ vw,
                              const float* __restrict__ vs, const float* __restrict__ vb,
                              __bf16* __restrict__ vT) {
  int i = blockIdx.x * blockDim.x + threadIdx.x;
  if (i >= 8 * 256 * 160) return;
  int n = i % 160, rest = i / 160, f = rest % 256, b = rest / 256;
  if (n >= 150) { vT[i] = (__bf16)0.f; return; }
  const float* cr = cc + (size_t)(b * 150 + n) * 512;
  float a = 0.f;
  for (int k = 0; k < 512; ++k) a += cr[k] * vw[k * 256 + f];
  a = a * vs[f] + vb[f];
  vT[i] = (__bf16)(a > 0.f ? a : 0.f);
}

// ---------------------------------------------------------------------------
// Fused main kernel: per block = 32 pixel rows, full chain in LDS.
// ---------------------------------------------------------------------------
__global__ __launch_bounds__(256) void k_oab_main(
    const float* __restrict__ x,
    const __bf16* __restrict__ qw0T, const float* __restrict__ qs0, const float* __restrict__ qb0,
    const __bf16* __restrict__ qw1T, const float* __restrict__ qs1, const float* __restrict__ qb1,
    const __bf16* __restrict__ kbf, const __bf16* __restrict__ vT,
    const __bf16* __restrict__ uwT, const float* __restrict__ us, const float* __restrict__ ub,
    float* __restrict__ out) {
  __shared__ __align__(64) char smem[53248];
  __bf16* q0s  = (__bf16*)smem;            // [32][256] bf16 (16KB)
  __bf16* q1s  = (__bf16*)(smem + 16384);  // [32][256] bf16 (16KB)
  float*  sim  = (float*)(smem + 32768);   // [32][160] f32  (20KB)
  __bf16* attn = (__bf16*)(smem + 16384);  // [32][160] bf16 (overlays q1s)
  __bf16* ys   = (__bf16*)smem;            // [32][256] bf16 (overlays q0s)

  const int tid = threadIdx.x;
  const int w = tid >> 5;
  const int l = tid & 31;
  const int lhi = l >> 4;   // which K-half this lane owns
  const int llo = l & 15;
  const int row_base = blockIdx.x * 32;
  const int batch = row_base >> 14;  // / (128*128)

  // ---- Stage 1: q0 = relu((x @ qw0)*qs0 + qb0) : [32,256], K=512 ----------
  {
    const int rt = w >> 2;             // 0: rows 0..15, 1: rows 16..31
    const int ct0 = (w & 3) * 4;       // 4 col tiles of 16 per wave
    const int m = llo + 16 * rt;
    const float* xrow = x + (size_t)(row_base + m) * 512;
    v8f acc[4] = {};
#pragma unroll 2
    for (int ks = 0; ks < 16; ++ks) {
      v16bf a = a_frag_from_f32(xrow, ks * 32 + lhi * 8);
      const int kb = ks * 32 + lhi * 16;
#pragma unroll
      for (int j = 0; j < 4; ++j) {
        v16bf b = b_frag(qw0T + (size_t)(llo + (ct0 + j) * 16) * 512 + kb);
        acc[j] = wmma_bf16(a, b, acc[j]);
      }
    }
#pragma unroll
    for (int j = 0; j < 4; ++j) {
      const int n = llo + (ct0 + j) * 16;
      const float s = qs0[n], bb = qb0[n];
#pragma unroll
      for (int r = 0; r < 8; ++r) {
        float vv = acc[j][r] * s + bb;
        vv = vv > 0.f ? vv : 0.f;
        q0s[(16 * rt + r + 8 * lhi) * 256 + n] = (__bf16)vv;
      }
    }
  }
  __syncthreads();

  // ---- Stage 2: q1 = relu((q0 @ qw1)*qs1 + qb1) : [32,256], K=256 ---------
  {
    const int rt = w >> 2;
    const int ct0 = (w & 3) * 4;
    const int m = llo + 16 * rt;
    v8f acc[4] = {};
#pragma unroll
    for (int ks = 0; ks < 8; ++ks) {
      v16bf a = a_frag_from_lds(q0s, m, 256, ks * 32 + lhi * 8);
      const int kb = ks * 32 + lhi * 16;
#pragma unroll
      for (int j = 0; j < 4; ++j) {
        v16bf b = b_frag(qw1T + (size_t)(llo + (ct0 + j) * 16) * 256 + kb);
        acc[j] = wmma_bf16(a, b, acc[j]);
      }
    }
    __syncthreads();  // ensure all q0s reads done before q1s writes? (distinct buffers, but keep order)
#pragma unroll
    for (int j = 0; j < 4; ++j) {
      const int n = llo + (ct0 + j) * 16;
      const float s = qs1[n], bb = qb1[n];
#pragma unroll
      for (int r = 0; r < 8; ++r) {
        float vv = acc[j][r] * s + bb;
        vv = vv > 0.f ? vv : 0.f;
        q1s[(16 * rt + r + 8 * lhi) * 256 + n] = (__bf16)vv;
      }
    }
  }
  __syncthreads();

  // ---- Stage 3: sim = (q1 @ k^T) / 16 : [32,160], K=256 -------------------
  {
    for (int t = w; t < 20; t += 8) {
      const int rt = t / 10, ct = t % 10;
      const int m = llo + 16 * rt;
      const __bf16* bbase = kbf + ((size_t)batch * 160 + llo + ct * 16) * 256;
      v8f acc = {};
#pragma unroll
      for (int ks = 0; ks < 8; ++ks) {
        v16bf a = a_frag_from_lds(q1s, m, 256, ks * 32 + lhi * 8);
        v16bf b = b_frag(bbase + ks * 32 + lhi * 16);
        acc = wmma_bf16(a, b, acc);
      }
      const int n = llo + ct * 16;
#pragma unroll
      for (int r = 0; r < 8; ++r)
        sim[(16 * rt + r + 8 * lhi) * 160 + n] = acc[r] * 0.0625f;
    }
  }
  __syncthreads();

  // ---- Stage 3.5: row softmax over 150 valid classes (pads -> 0) ----------
  {
#pragma unroll
    for (int i = 0; i < 4; ++i) {
      const int row = w * 4 + i;
      const float* srow = sim + row * 160;
      float mx = -3.0e38f;
      for (int c = l; c < 150; c += 32) mx = fmaxf(mx, srow[c]);
#pragma unroll
      for (int off = 16; off >= 1; off >>= 1) mx = fmaxf(mx, __shfl_xor(mx, off, 32));
      float e[5];
      float sum = 0.f;
#pragma unroll
      for (int j = 0; j < 5; ++j) {
        const int c = l + j * 32;
        float ev = 0.f;
        if (c < 150) ev = __expf(srow[c] - mx);
        e[j] = ev;
        sum += ev;
      }
#pragma unroll
      for (int off = 16; off >= 1; off >>= 1) sum += __shfl_xor(sum, off, 32);
      const float inv = 1.0f / sum;
      __bf16* arow = attn + row * 160;
#pragma unroll
      for (int j = 0; j < 5; ++j) arow[l + j * 32] = (__bf16)(e[j] * inv);
    }
  }
  __syncthreads();

  // ---- Stage 4: y = attn @ v : [32,256], K=160 ----------------------------
  {
    const int rt = w >> 2;
    const int ct0 = (w & 3) * 4;
    const int m = llo + 16 * rt;
    v8f acc[4] = {};
#pragma unroll
    for (int ks = 0; ks < 5; ++ks) {
      v16bf a = a_frag_from_lds(attn, m, 160, ks * 32 + lhi * 8);
      const int kb = ks * 32 + lhi * 16;
#pragma unroll
      for (int j = 0; j < 4; ++j) {
        v16bf b = b_frag(vT + ((size_t)batch * 256 + llo + (ct0 + j) * 16) * 160 + kb);
        acc[j] = wmma_bf16(a, b, acc[j]);
      }
    }
    __syncthreads();  // attn (and thus q1s slot) reads finished before ys writes to q0s slot
#pragma unroll
    for (int j = 0; j < 4; ++j) {
      const int n = llo + (ct0 + j) * 16;
#pragma unroll
      for (int r = 0; r < 8; ++r)
        ys[(16 * rt + r + 8 * lhi) * 256 + n] = (__bf16)acc[j][r];
    }
  }
  __syncthreads();

  // ---- Stage 5: out = relu((y @ uw)*us + ub) : [32,512], K=256 ------------
  {
    const int rt = w >> 2;
    const int ct0 = (w & 3) * 8;  // 8 col tiles of 16 per wave -> 512 cols
    const int m = llo + 16 * rt;
    v8f acc[8] = {};
#pragma unroll
    for (int ks = 0; ks < 8; ++ks) {
      v16bf a = a_frag_from_lds(ys, m, 256, ks * 32 + lhi * 8);
      const int kb = ks * 32 + lhi * 16;
#pragma unroll
      for (int j = 0; j < 8; ++j) {
        v16bf b = b_frag(uwT + (size_t)(llo + (ct0 + j) * 16) * 256 + kb);
        acc[j] = wmma_bf16(a, b, acc[j]);
      }
    }
#pragma unroll
    for (int j = 0; j < 8; ++j) {
      const int n = llo + (ct0 + j) * 16;
      const float s = us[n], bb = ub[n];
#pragma unroll
      for (int r = 0; r < 8; ++r) {
        const int mm = 16 * rt + r + 8 * lhi;
        float vv = acc[j][r] * s + bb;
        vv = vv > 0.f ? vv : 0.f;
        out[(size_t)(row_base + mm) * 512 + n] = vv;
      }
    }
  }
}

// ---------------------------------------------------------------------------
// Launch
// ---------------------------------------------------------------------------
extern "C" void kernel_launch(void* const* d_in, const int* in_sizes, int n_in,
                              void* d_out, int out_size, void* d_ws, size_t ws_size,
                              hipStream_t stream) {
  const float* x   = (const float*)d_in[0];
  const float* cc  = (const float*)d_in[1];
  const float* qw0 = (const float*)d_in[2];
  const float* qs0 = (const float*)d_in[3];
  const float* qb0 = (const float*)d_in[4];
  const float* qw1 = (const float*)d_in[5];
  const float* qs1 = (const float*)d_in[6];
  const float* qb1 = (const float*)d_in[7];
  const float* kw0 = (const float*)d_in[8];
  const float* ks0 = (const float*)d_in[9];
  const float* kb0 = (const float*)d_in[10];
  const float* kw1 = (const float*)d_in[11];
  const float* ks1 = (const float*)d_in[12];
  const float* kb1 = (const float*)d_in[13];
  const float* vw  = (const float*)d_in[14];
  const float* vs  = (const float*)d_in[15];
  const float* vb  = (const float*)d_in[16];
  const float* uw  = (const float*)d_in[17];
  const float* us  = (const float*)d_in[18];
  const float* ub  = (const float*)d_in[19];
  float* out = (float*)d_out;

  char* ws = (char*)d_ws;
  __bf16* qw0T = (__bf16*)(ws);             // [256][512] bf16  (256KB)
  __bf16* qw1T = (__bf16*)(ws + 262144);    // [256][256] bf16  (128KB)
  __bf16* uwT  = (__bf16*)(ws + 393216);    // [512][256] bf16  (256KB)
  __bf16* kbf  = (__bf16*)(ws + 655360);    // [8][160][256] bf16 (640KB)
  __bf16* vT   = (__bf16*)(ws + 1310720);   // [8][256][160] bf16 (640KB)
  float*  t1   = (float*)(ws + 1966080);    // [8*150][256] f32 (1.2MB)

  // weight transposes -> bf16 [n][k]
  k_transpose_bf16<<<(512 * 256 + 255) / 256, 256, 0, stream>>>(qw0, qw0T, 512, 256);
  k_transpose_bf16<<<(256 * 256 + 255) / 256, 256, 0, stream>>>(qw1, qw1T, 256, 256);
  k_transpose_bf16<<<(256 * 512 + 255) / 256, 256, 0, stream>>>(uw, uwT, 256, 512);

  // class key/value path (tiny)
  k_cbr_naive<<<(1200 * 256 + 255) / 256, 256, 0, stream>>>(cc, kw0, ks0, kb0, t1, 1200, 512, 256);
  k_class_key<<<(8 * 160 * 256 + 255) / 256, 256, 0, stream>>>(t1, kw1, ks1, kb1, kbf);
  k_class_value<<<(8 * 256 * 160 + 255) / 256, 256, 0, stream>>>(cc, vw, vs, vb, vT);

  // fused pixel pipeline: 131072 rows / 32 per block
  k_oab_main<<<4096, 256, 0, stream>>>(x, qw0T, qs0, qb0, qw1T, qs1, qb1,
                                       kbf, vT, uwT, us, ub, out);
}
